// CGI_MH_30124900614631
// MI455X (gfx1250) — matrix operand order repaired and checked by reading, not compiled, not run on hardware
//
#include <hip/hip_runtime.h>
#include <stdint.h>

// Problem constants
#define BB 16
#define CC 1024
#define NN 1024      // H*W
#define TT 77
#define TP 96        // padded T (3 K-chunks of 32)
#define NH 8
#define HD 128

typedef __attribute__((ext_vector_type(16))) __bf16 v16bf;
typedef __attribute__((ext_vector_type(8)))  __bf16 v8bf;
typedef __attribute__((ext_vector_type(2)))  __bf16 v2bf;
typedef __attribute__((ext_vector_type(8)))  float  v8f;

__device__ __forceinline__ v8f wmma_bf16(v16bf a, v16bf b, v8f c) {
  return __builtin_amdgcn_wmma_f32_16x16x32_bf16(false, a, false, b, (short)0, c,
                                                 false, false);
}

// A fragment 16x32 (MxK), source layout [m][k] with row stride (mult of 8).
__device__ __forceinline__ v16bf frag_a_ld(const __bf16* p, int stride, int lane) {
  const int m  = lane & 15;
  const int kh = (lane >> 4) * 8;
  const __bf16* q = p + m * stride + kh;
  v8bf lo = *(const v8bf*)(q);
  v8bf hi = *(const v8bf*)(q + 16);
  return __builtin_shufflevector(lo, hi, 0, 1, 2, 3, 4, 5, 6, 7, 8, 9, 10, 11,
                                 12, 13, 14, 15);
}

// B fragment 32x16 (KxN), source layout [n][k] with row stride (mult of 8).
__device__ __forceinline__ v16bf frag_b_ld(const __bf16* p, int stride, int lane) {
  const int n  = lane & 15;
  const int kg = (lane >> 4) * 8;
  const __bf16* q = p + n * stride + kg;
  v8bf lo = *(const v8bf*)(q);
  v8bf hi = *(const v8bf*)(q + 16);
  return __builtin_shufflevector(lo, hi, 0, 1, 2, 3, 4, 5, 6, 7, 8, 9, 10, 11,
                                 12, 13, 14, 15);
}

__device__ __forceinline__ v8f zero8() {
  v8f z;
#pragma unroll
  for (int i = 0; i < 8; ++i) z[i] = 0.0f;
  return z;
}

// ------------------------- Tensor Data Mover helpers ------------------------
#if defined(__has_builtin)
#if __has_builtin(__builtin_amdgcn_tensor_load_to_lds)
#define HAVE_TDM 1
#endif
#endif

#ifdef HAVE_TDM
typedef __attribute__((ext_vector_type(4))) unsigned tdm_u4;
typedef __attribute__((ext_vector_type(8))) int      tdm_i8;
typedef __attribute__((ext_vector_type(4))) int      tdm_i4;

// 2-D bf16 tile load: dim0 contiguous elements per row, dim1 rows; optional
// LDS row padding (pad_interval/pad_amount per CDNA5 D# encoding, minus-1).
__device__ __forceinline__ void tdm_load_2d_bf16(unsigned lds_byte_addr,
                                                 const void* gaddr,
                                                 unsigned dim0, unsigned dim1,
                                                 unsigned pad_en,
                                                 unsigned pad_interval,
                                                 unsigned pad_amount) {
  const unsigned long long ga = (unsigned long long)gaddr;
  tdm_u4 g0;
  g0[0] = 1u;                                            // count=1 (valid)
  g0[1] = lds_byte_addr;                                 // lds_addr
  g0[2] = (unsigned)(ga & 0xFFFFFFFFu);                  // global_addr[31:0]
  g0[3] = (unsigned)((ga >> 32) & 0x01FFFFFFu) | (2u << 30);  // +type=2
  tdm_i8 g1;
  g1[0] = (int)((1u << 16) |                             // data_size=2B
                (pad_en << 20) | (pad_interval << 22) | (pad_amount << 25));
  g1[1] = (int)((dim0 & 0xFFFFu) << 16);                 // tensor_dim0 lo16
  g1[2] = (int)(((dim0 >> 16) & 0xFFFFu) | ((dim1 & 0xFFFFu) << 16));
  g1[3] = (int)(((dim1 >> 16) & 0xFFFFu) | ((dim0 & 0xFFFFu) << 16)); // tile0
  g1[4] = (int)(dim1 & 0xFFFFu);                         // tile1 (tile2=0)
  g1[5] = (int)dim0;                                     // dim0_stride lo32
  g1[6] = 0;
  g1[7] = 0;
  tdm_i4 z4 = {0, 0, 0, 0};
  tdm_i8 z8 = {0, 0, 0, 0, 0, 0, 0, 0};
  __builtin_amdgcn_tensor_load_to_lds(g0, g1, z4, z4, z8, 0);
}
#endif

// ---------------------------------------------------------------------------
// Kernel 1: Q projection. q[b,h,n,d] = sum_c Wq[h*128+d,c]*Vx[b,c,n] + bq
// WMMA: M=n, N=d, K=c. Each wave owns TWO 16-row A tiles (32 n rows) so each
// B fragment feeds two WMMAs. grid (dtile=8, ntile256=4, b=16), 256 threads.
// ---------------------------------------------------------------------------
__global__ __launch_bounds__(256) void qproj_kernel(
    const float* __restrict__ Vx, const float* __restrict__ Wq,
    const float* __restrict__ bq, __bf16* __restrict__ q_ws) {
  __shared__ alignas(16) __bf16 lds_x[256 * 40];  // [n][c] stride 40
  __shared__ alignas(16) __bf16 lds_w[128 * 40];  // [d][c] stride 40
  const int tid = threadIdx.x, lane = tid & 31, wave = tid >> 5;
  const int dbase = blockIdx.x * 128;
  const int nbase = blockIdx.y * 256;
  const int b     = blockIdx.z;

  v8f acc0[8], acc1[8];
#pragma unroll
  for (int j = 0; j < 8; ++j) { acc0[j] = zero8(); acc1[j] = zero8(); }

  for (int c0 = 0; c0 < CC; c0 += 32) {
    // Vx chunk (32c x 256n): coalesced float2 reads, scatter-transpose [n][c]
#pragma unroll
    for (int i = 0; i < 16; ++i) {
      int idx = tid + 256 * i;              // 4096 float2
      int c = idx >> 7, n0 = (idx & 127) * 2;
      const float2 v =
          *(const float2*)(Vx + ((size_t)b * CC + c0 + c) * NN + nbase + n0);
      lds_x[n0 * 40 + c]       = (__bf16)v.x;
      lds_x[(n0 + 1) * 40 + c] = (__bf16)v.y;
    }
    // Wq chunk: natural row-major [d][c]
#pragma unroll
    for (int i = 0; i < 8; ++i) {
      int idx = tid + 256 * i;              // 2048 float2
      int dd = idx >> 4, cp = (idx & 15) * 2;
      const float2 v =
          *(const float2*)(Wq + (size_t)(dbase + dd) * CC + c0 + cp);
      v2bf w; w[0] = (__bf16)v.x; w[1] = (__bf16)v.y;
      *(v2bf*)(lds_w + dd * 40 + cp) = w;
    }
    __syncthreads();
    v16bf a0 = frag_a_ld(lds_x + (wave * 32) * 40, 40, lane);
    v16bf a1 = frag_a_ld(lds_x + (wave * 32 + 16) * 40, 40, lane);
#pragma unroll
    for (int j = 0; j < 8; ++j) {
      v16bf bf = frag_b_ld(lds_w + j * 16 * 40, 40, lane);
      acc0[j] = wmma_bf16(a0, bf, acc0[j]);
      acc1[j] = wmma_bf16(a1, bf, acc1[j]);
    }
    __syncthreads();
  }

  const int col = lane & 15, half = lane >> 4;
  const int h = blockIdx.x;
#pragma unroll
  for (int j = 0; j < 8; ++j) {
    const int dloc = j * 16 + col;
    const float bv = bq[dbase + dloc];
#pragma unroll
    for (int r = 0; r < 8; ++r) {
      int n0 = nbase + wave * 32 + r + 8 * half;
      q_ws[(((size_t)b * NH + h) * NN + n0) * HD + dloc] =
          (__bf16)(acc0[j][r] + bv);
      q_ws[(((size_t)b * NH + h) * NN + n0 + 16) * HD + dloc] =
          (__bf16)(acc1[j][r] + bv);
    }
  }
}

// ---------------------------------------------------------------------------
// Kernel 2: KV projection. kv[b,d2,t] = sum_c Wkv[d2,c]*Tx[b,c,t] + bkv
// WMMA: M=d2, N=t, K=c. k region [b][h][t][d] (t-major), v region [b][h][d][t]
// grid (d2tile=16, b=16).
// ---------------------------------------------------------------------------
__global__ __launch_bounds__(256) void kvproj_kernel(
    const float* __restrict__ Tx, const float* __restrict__ Wkv,
    const float* __restrict__ bkv, __bf16* __restrict__ kv_ws) {
  __shared__ alignas(16) __bf16 lds_w[128 * 40];  // [d2][c] stride 40
  __shared__ alignas(16) __bf16 lds_t[80 * 40];   // [t][c] stride 40
  const int tid = threadIdx.x, lane = tid & 31, wave = tid >> 5;
  const int d2base = blockIdx.x * 128;
  const int b      = blockIdx.y;

  v8f acc[5];
#pragma unroll
  for (int j = 0; j < 5; ++j) acc[j] = zero8();

  for (int c0 = 0; c0 < CC; c0 += 32) {
#pragma unroll
    for (int i = 0; i < 8; ++i) {
      int idx = tid + 256 * i;
      int dd = idx >> 4, cp = (idx & 15) * 2;
      const float2 v =
          *(const float2*)(Wkv + (size_t)(d2base + dd) * CC + c0 + cp);
      v2bf w; w[0] = (__bf16)v.x; w[1] = (__bf16)v.y;
      *(v2bf*)(lds_w + dd * 40 + cp) = w;
    }
#pragma unroll
    for (int i = 0; i < 10; ++i) {          // 2560 elems (t padded to 80)
      int idx = tid + 256 * i;
      int cc = idx / 80, t = idx % 80;
      float v = (t < TT) ? Tx[((size_t)b * CC + c0 + cc) * TT + t] : 0.0f;
      lds_t[t * 40 + cc] = (__bf16)v;
    }
    __syncthreads();
    v16bf a = frag_a_ld(lds_w + wave * 16 * 40, 40, lane);
#pragma unroll
    for (int tt = 0; tt < 5; ++tt) {
      v16bf bf = frag_b_ld(lds_t + tt * 16 * 40, 40, lane);
      acc[tt] = wmma_bf16(a, bf, acc[tt]);
    }
    __syncthreads();
  }

  const int col = lane & 15, half = lane >> 4;
#pragma unroll
  for (int r = 0; r < 8; ++r) {
    const int d2g = d2base + wave * 16 + r + 8 * half;
    const float bv = bkv[d2g];
    const int h = d2g >> 8, rem = d2g & 255;
    const int which = rem >> 7, dl = rem & 127;
    const size_t base = ((size_t)(b * NH + h) * 2 + which) * (HD * TP);
#pragma unroll
    for (int tt = 0; tt < 5; ++tt) {
      int t = tt * 16 + col;
      float val = acc[tt][r] + bv;
      if (which == 0)
        kv_ws[base + (size_t)t * HD + dl] = (__bf16)val;   // k: [t][d]
      else
        kv_ws[base + (size_t)dl * TP + t] = (__bf16)val;   // v: [d][t]
    }
  }
}

// ---------------------------------------------------------------------------
// Kernel 3: attention. S = q.k/sqrt(hd); softmax over t; O = v.attn^T
// k/v staged via Tensor Data Mover (wave 0 issues 2 descriptors; row padding
// gives k LDS stride 136). Dynamic LDS: [0, 26624) k->attn, [26624, ...) v.
// grid (ntile=8, h=8, b=16), 256 threads.
// ---------------------------------------------------------------------------
#define KSTR 136             // k LDS row stride ([t][d], 128 + 8 pad)
#define VSTR TP              // v LDS row stride ([d][t], 96, unpadded)
#define ASTR 104             // attn LDS row stride ([n][t], 96 + 8 pad)
#define AREGION 26624        // bytes: max(96*KSTR, 128*ASTR)*2
#define SMEM_BYTES (AREGION + 128 * VSTR * 2)
__global__ __launch_bounds__(256) void attn_kernel(
    const __bf16* __restrict__ q_ws, const __bf16* __restrict__ kv_ws,
    __bf16* __restrict__ o_ws) {
  extern __shared__ __bf16 smem[];
  __bf16* smA = smem;                                  // k then attn
  __bf16* smV = smem + AREGION / 2;                    // v [d][t]
  const int tid = threadIdx.x, lane = tid & 31, wave = tid >> 5;
  const int nbase = blockIdx.x * 128;
  const int h = blockIdx.y, b = blockIdx.z;

  const size_t kb = ((size_t)(b * NH + h) * 2 + 0) * (HD * TP);
  const size_t vb = ((size_t)(b * NH + h) * 2 + 1) * (HD * TP);
#ifdef HAVE_TDM
  if (wave == 0) {
    const unsigned lds0 = __builtin_amdgcn_groupstaticsize();
    // k: 96 rows x 128 elems, pad 4 DWORDs (enc 3) per 64 DWORDs (enc 5)
    tdm_load_2d_bf16(lds0, kv_ws + kb, HD, TP, 1u, 5u, 3u);
    // v: 128 rows x 96 elems, contiguous
    tdm_load_2d_bf16(lds0 + AREGION, kv_ws + vb, TP, HD, 0u, 0u, 0u);
    __builtin_amdgcn_s_wait_tensorcnt(0);
  }
#else
  {
    const uint4* srck = (const uint4*)(kv_ws + kb);
    const uint4* srcv = (const uint4*)(kv_ws + vb);
    uint4* dk = (uint4*)smA;
    uint4* dv = (uint4*)smV;
#pragma unroll
    for (int i = 0; i < 6; ++i) {
      int idx = tid + 256 * i;              // 1536 uint4 each
      int kr = idx >> 4, kc = idx & 15;
      dk[kr * (KSTR / 8) + kc] = srck[idx];
      dv[idx] = srcv[idx];                  // v rows contiguous (stride 96)
    }
  }
#endif
  __syncthreads();

  // ---- S = q * k : M=n(16/wave), N=t(5 tiles), K=d(4x32) ----
  v8f s[5];
#pragma unroll
  for (int t = 0; t < 5; ++t) s[t] = zero8();
  const __bf16* qrow =
      q_ws + (((size_t)b * NH + h) * NN + nbase + wave * 16) * HD;
#pragma unroll
  for (int kc = 0; kc < 4; ++kc) {
    v16bf a = frag_a_ld(qrow + kc * 32, HD, lane);   // global b128 loads
#pragma unroll
    for (int tt = 0; tt < 5; ++tt) {
      v16bf bf = frag_b_ld(smA + tt * 16 * KSTR + kc * 32, KSTR, lane);
      s[tt] = wmma_bf16(a, bf, s[tt]);
    }
  }

  // ---- softmax over t (row lives in one VGPR slot across a 16-lane half) --
  const int col = lane & 15, half = lane >> 4;
  const float scale = 0.08838834764831845f;  // 1/sqrt(128)
  float attn[5][8];
#pragma unroll
  for (int r = 0; r < 8; ++r) {
    float vals[5];
    float m = -3.0e30f;
#pragma unroll
    for (int tt = 0; tt < 5; ++tt) {
      int t = tt * 16 + col;
      vals[tt] = (t < TT) ? s[tt][r] * scale : -3.0e30f;
      m = fmaxf(m, vals[tt]);
    }
#pragma unroll
    for (int off = 8; off >= 1; off >>= 1)
      m = fmaxf(m, __shfl_xor(m, off, 16));
    float sum = 0.0f;
#pragma unroll
    for (int tt = 0; tt < 5; ++tt) {
      int t = tt * 16 + col;
      float e = (t < TT) ? __expf(vals[tt] - m) : 0.0f;
      vals[tt] = e;
      sum += e;
    }
#pragma unroll
    for (int off = 8; off >= 1; off >>= 1) sum += __shfl_xor(sum, off, 16);
    float inv = 1.0f / sum;
#pragma unroll
    for (int tt = 0; tt < 5; ++tt) attn[tt][r] = vals[tt] * inv;
  }
  __syncthreads();  // all waves done reading k from smA

  // write attn [n][t] stride ASTR (+ zero pad t=80..95)
#pragma unroll
  for (int r = 0; r < 8; ++r) {
    int row = wave * 16 + r + 8 * half;
#pragma unroll
    for (int tt = 0; tt < 5; ++tt)
      smA[row * ASTR + tt * 16 + col] = (__bf16)attn[tt][r];
    smA[row * ASTR + 80 + col] = (__bf16)0.0f;
  }
  __syncthreads();

  // ---- O = v * attn^T : M=d(16/wave), N=n(8 tiles), K=t(3x32) ----
  v8f o[8];
#pragma unroll
  for (int j = 0; j < 8; ++j) o[j] = zero8();
#pragma unroll
  for (int kc = 0; kc < 3; ++kc) {
    v16bf a = frag_a_ld(smV + wave * 16 * VSTR + kc * 32, VSTR, lane);
#pragma unroll
    for (int j = 0; j < 8; ++j) {
      v16bf bf = frag_b_ld(smA + j * 16 * ASTR + kc * 32, ASTR, lane);
      o[j] = wmma_bf16(a, bf, o[j]);
    }
  }
#pragma unroll
  for (int r = 0; r < 8; ++r) {
    int c = h * HD + wave * 16 + r + 8 * half;
#pragma unroll
    for (int j = 0; j < 8; ++j) {
      int n = nbase + j * 16 + col;
      o_ws[((size_t)b * CC + c) * NN + n] = (__bf16)o[j][r];
    }
  }
}

// ---------------------------------------------------------------------------
// Kernel 4: output projection. out[b,d,n] = sum_c Wp[d,c]*O[b,c,n] + bp (f32)
// WMMA: M=d, N=n, K=c. Each wave owns TWO 16-row A tiles (32 d rows).
// grid (dtile256=4, ntile=8, b=16).
// ---------------------------------------------------------------------------
__global__ __launch_bounds__(256) void oproj_kernel(
    const __bf16* __restrict__ o_ws, const float* __restrict__ Wp,
    const float* __restrict__ bp, float* __restrict__ out) {
  __shared__ alignas(16) __bf16 lds_w[256 * 40];  // [d][c] stride 40
  __shared__ alignas(16) __bf16 lds_o[128 * 40];  // [n][c] stride 40
  const int tid = threadIdx.x, lane = tid & 31, wave = tid >> 5;
  const int dbase = blockIdx.x * 256;
  const int nbase = blockIdx.y * 128;
  const int b     = blockIdx.z;

  v8f acc0[8], acc1[8];
#pragma unroll
  for (int j = 0; j < 8; ++j) { acc0[j] = zero8(); acc1[j] = zero8(); }

  for (int c0 = 0; c0 < CC; c0 += 32) {
#pragma unroll
    for (int i = 0; i < 16; ++i) {          // Wp: 4096 float2 (256 d rows)
      int idx = tid + 256 * i;
      int dd = idx >> 4, cp = (idx & 15) * 2;
      const float2 v =
          *(const float2*)(Wp + (size_t)(dbase + dd) * CC + c0 + cp);
      v2bf w; w[0] = (__bf16)v.x; w[1] = (__bf16)v.y;
      *(v2bf*)(lds_w + dd * 40 + cp) = w;
    }
#pragma unroll
    for (int i = 0; i < 8; ++i) {           // O chunk: 2048 bf16-pairs
      int idx = tid + 256 * i;
      int c = idx >> 6, n0 = (idx & 63) * 2;
      v2bf v =
          *(const v2bf*)(o_ws + ((size_t)b * CC + c0 + c) * NN + nbase + n0);
      lds_o[n0 * 40 + c]       = v[0];
      lds_o[(n0 + 1) * 40 + c] = v[1];
    }
    __syncthreads();
    v16bf a0 = frag_a_ld(lds_w + (wave * 32) * 40, 40, lane);
    v16bf a1 = frag_a_ld(lds_w + (wave * 32 + 16) * 40, 40, lane);
#pragma unroll
    for (int j = 0; j < 8; ++j) {
      v16bf bf = frag_b_ld(lds_o + j * 16 * 40, 40, lane);
      acc0[j] = wmma_bf16(a0, bf, acc0[j]);
      acc1[j] = wmma_bf16(a1, bf, acc1[j]);
    }
    __syncthreads();
  }

  const int col = lane & 15, half = lane >> 4;
#pragma unroll
  for (int r = 0; r < 8; ++r) {
    const int dg0 = dbase + wave * 32 + r + 8 * half;
    const float bv0 = bp[dg0];
    const float bv1 = bp[dg0 + 16];
#pragma unroll
    for (int j = 0; j < 8; ++j) {
      int n = nbase + j * 16 + col;
      out[((size_t)b * CC + dg0) * NN + n]      = acc0[j][r] + bv0;
      out[((size_t)b * CC + dg0 + 16) * NN + n] = acc1[j][r] + bv1;
    }
  }
}

// ---------------------------------------------------------------------------
extern "C" void kernel_launch(void* const* d_in, const int* in_sizes, int n_in,
                              void* d_out, int out_size, void* d_ws,
                              size_t ws_size, hipStream_t stream) {
  const float* Vx  = (const float*)d_in[0];
  const float* Tx  = (const float*)d_in[1];
  const float* Wq  = (const float*)d_in[2];
  const float* bq  = (const float*)d_in[3];
  const float* Wkv = (const float*)d_in[4];
  const float* bkv = (const float*)d_in[5];
  const float* Wp  = (const float*)d_in[6];
  const float* bp  = (const float*)d_in[7];
  float* out = (float*)d_out;

  char* ws = (char*)d_ws;
  const size_t q_bytes  = (size_t)BB * NH * NN * HD * sizeof(__bf16);  // 32 MB
  const size_t kv_bytes = (size_t)BB * NH * 2 * HD * TP * sizeof(__bf16);
  __bf16* q_ws  = (__bf16*)ws;
  __bf16* kv_ws = (__bf16*)(ws + q_bytes);
  __bf16* o_ws  = (__bf16*)(ws + q_bytes + kv_bytes);

  (void)hipMemsetAsync(kv_ws, 0, kv_bytes, stream);  // zero T-pad region

  qproj_kernel<<<dim3(8, 4, BB), 256, 0, stream>>>(Vx, Wq, bq, q_ws);
  kvproj_kernel<<<dim3(16, BB), 256, 0, stream>>>(Tx, Wkv, bkv, kv_ws);
  attn_kernel<<<dim3(8, NH, BB), 256, SMEM_BYTES, stream>>>(q_ws, kv_ws, o_ws);
  oproj_kernel<<<dim3(4, 8, BB), 256, 0, stream>>>(o_ws, Wp, bp, out);
}
